// Starcoder2Attention_14645838480144
// MI455X (gfx1250) — compile-verified, matrix operand order checked by reading, not run on hardware
//
#include <hip/hip_runtime.h>
#include <hip/hip_bf16.h>

// ---------------------------------------------------------------------------
// Starcoder2 attention block for MI455X (gfx1250, wave32, WMMA + TDM).
//   H=36 q-heads, KV=4 kv-heads (GQA group 9), D=128, HID=4608, T=2048,
//   sliding window 1024.  fp32 in/out, bf16 WMMA (f32 accumulate) inside.
// ---------------------------------------------------------------------------

#define TT    2048
#define HEADS 36
#define KVH   4
#define DH    128
#define HID   4608
#define NQKV  (HEADS * DH + 2 * KVH * DH) // 5632

typedef __attribute__((ext_vector_type(16))) __bf16 v16bf;
typedef __attribute__((ext_vector_type(8)))  float  v8f;
typedef unsigned int su4 __attribute__((ext_vector_type(4)));
typedef unsigned int su8 __attribute__((ext_vector_type(8)));

struct alignas(16) U4  { unsigned int x, y, z, w; };
struct alignas(16) F4s { float x, y, z, w; };
struct alignas(32) F8  { U4 lo, hi; };

static __device__ __forceinline__ v8f vzero() {
  v8f z = {0.f, 0.f, 0.f, 0.f, 0.f, 0.f, 0.f, 0.f};
  return z;
}

// fp32 -> bf16 (round to nearest even)
static __device__ __forceinline__ unsigned short f2bf(float f) {
  unsigned int u = __builtin_bit_cast(unsigned int, f);
  u += 0x7fffu + ((u >> 16) & 1u);
  return (unsigned short)(u >> 16);
}
static __device__ __forceinline__ unsigned int pk2(float a, float b) {
  return (unsigned int)f2bf(a) | ((unsigned int)f2bf(b) << 16);
}

// DPP16 row rotate (rows are 16 lanes on wave32 == one C-fragment group).
// Rotate-reduce with n = 1,2,4,8 leaves every lane holding the row reduction.
#define DPP_ROR_F(x, n)                                                      \
  __builtin_bit_cast(float, __builtin_amdgcn_update_dpp(                     \
      0, __builtin_bit_cast(int, (x)), (0x120 | (n)), 0xf, 0xf, true))

static __device__ __forceinline__ float rowmax16(float x) {
  x = fmaxf(x, DPP_ROR_F(x, 1));
  x = fmaxf(x, DPP_ROR_F(x, 2));
  x = fmaxf(x, DPP_ROR_F(x, 4));
  x = fmaxf(x, DPP_ROR_F(x, 8));
  return x;
}
static __device__ __forceinline__ float rowsum16(float x) {
  x += DPP_ROR_F(x, 1);
  x += DPP_ROR_F(x, 2);
  x += DPP_ROR_F(x, 4);
  x += DPP_ROR_F(x, 8);
  return x;
}

// ---------------------------------------------------------------------------
// Tensor Data Mover: async 2D bf16 tile load Global -> LDS.
// Builds the D# per CDNA5 ISA (sect. 8) in SGPRs:
//   group0: count=1 | lds_addr | global_addr(57b) | type=2
//   group1: data_size=2B, tensor_dim0/1 == tile_dim0/1 (tile always in
//           range -> no OOB clamping needed), tensor_dim0_stride = row pitch.
// LDS fill order matches row-major [tile_d1][tile_d0] bf16.
// ---------------------------------------------------------------------------
static __device__ __forceinline__ void tdm_load_2d_bf16(
    unsigned lds_addr, const void* gptr, unsigned tile_d0 /*elems*/,
    unsigned tile_d1 /*rows*/, unsigned row_stride_elems) {
  unsigned long long ga = (unsigned long long)(uintptr_t)gptr;
  su4 g0;
  g0[0] = 1u;                 // count=1, user descriptor, no gather
  g0[1] = lds_addr;           // LDS byte address
  g0[2] = (unsigned)ga;       // global_addr[31:0]
  g0[3] = (unsigned)((ga >> 32) & 0x1ffffffu) | (2u << 30);  // addr hi | type=2
  su8 g1;
  g1[0] = (1u << 16);                                   // data_size = 2 bytes
  g1[1] = (tile_d0 & 0xffffu) << 16;                    // tensor_dim0[15:0]
  g1[2] = (tile_d0 >> 16) | ((tile_d1 & 0xffffu) << 16);// td0 hi | td1 lo
  g1[3] = (tile_d1 >> 16) | (tile_d0 << 16);            // td1 hi | tile_dim0
  g1[4] = tile_d1 & 0xffffu;                            // tile_dim1, tile_dim2=0
  g1[5] = row_stride_elems;                             // tensor_dim0_stride lo
  g1[6] = 0u;                                           // stride hi | dim1_stride lo
  g1[7] = 0u;
  asm volatile("tensor_load_to_lds %0, %1" ::"s"(g0), "s"(g1) : "memory");
}

static __device__ __forceinline__ void wait_tensorcnt0() {
#if defined(__has_builtin) && __has_builtin(__builtin_amdgcn_s_wait_tensorcnt)
  __builtin_amdgcn_s_wait_tensorcnt(0);
#else
  asm volatile("s_wait_tensorcnt 0x0" ::: "memory");
#endif
}

// Load one 16x32 bf16 WMMA A/B fragment from a row-major [row][K] tile.
// Per ISA 7.12.2: lane&15 = row (A: M, B: N); lane>=16 holds K 8..15 / 24..31.
static __device__ __forceinline__ v16bf ld_frag(const unsigned short* base,
                                                int stride, int r0, int k0,
                                                int lane) {
  const int row = r0 + (lane & 15);
  const int kk  = k0 + ((lane >> 4) << 3);
  const unsigned short* p = base + row * stride + kk;
  F8 f;
  f.lo = *reinterpret_cast<const U4*>(p);
  f.hi = *reinterpret_cast<const U4*>(p + 16);
  return __builtin_bit_cast(v16bf, f);
}

static __device__ __forceinline__ v8f wmma_bf16(v16bf a, v16bf b, v8f c) {
  // v_wmma_f32_16x16x32_bf16
  return __builtin_amdgcn_wmma_f32_16x16x32_bf16(false, a, false, b,
                                                 (short)0, c, false, false);
}

// ---------------------------------------------------------------------------
// GEMM: C[M][N] = A[M][K] * B^T + bias, B stored as weight rows [N][K]
// (the contiguous-K layout the WMMA B fragment wants).
// A fp32 (ABF16=false, staged+converted via VGPRs) or bf16 (ABF16=true,
// staged by the Tensor Data Mover with zero VGPR traffic).
// Block tile 128x128x32, 256 threads = 8 waves (4Mx2N), each wave 32x64.
// ---------------------------------------------------------------------------
template <bool ABF16>
__global__ __launch_bounds__(256) void gemm_rowB_bias(
    const void* __restrict__ Ap, const float* __restrict__ Bw,
    const float* __restrict__ bias, float* __restrict__ C, int M, int N,
    int K) {
  __shared__ unsigned short As[128 * 32];
  __shared__ unsigned short Bs[128 * 32];

  const int tid  = threadIdx.x;
  const int lane = tid & 31;
  const int wave = tid >> 5;
  const int wm   = (wave & 3) * 32;
  const int wn   = (wave >> 2) * 64;
  const int bm   = blockIdx.y * 128;
  const int bn   = blockIdx.x * 128;

  v8f acc[2][4];
  for (int i = 0; i < 2; ++i)
    for (int j = 0; j < 4; ++j) acc[i][j] = vzero();

  for (int k0 = 0; k0 < K; k0 += 32) {
    // ---- stage A tile (128x32) into LDS as bf16 ----
    if (ABF16) {
      if (wave == 0)  // one TDM descriptor covers the whole tile
        tdm_load_2d_bf16((unsigned)(uintptr_t)(void*)As,
                         (const unsigned short*)Ap + (size_t)bm * K + k0,
                         /*tile_d0=*/32, /*tile_d1=*/128,
                         /*row_stride=*/(unsigned)K);
    } else {
      const float* A = (const float*)Ap;
      int row = tid >> 1, cb = (tid & 1) * 16;
      const F4s* src = (const F4s*)(A + (size_t)(bm + row) * K + k0 + cb);
      F4s f0 = src[0], f1 = src[1], f2 = src[2], f3 = src[3];
      U4 d0 = {pk2(f0.x, f0.y), pk2(f0.z, f0.w), pk2(f1.x, f1.y),
               pk2(f1.z, f1.w)};
      U4 d1 = {pk2(f2.x, f2.y), pk2(f2.z, f2.w), pk2(f3.x, f3.y),
               pk2(f3.z, f3.w)};
      *reinterpret_cast<U4*>(&As[row * 32 + cb]) = d0;
      *reinterpret_cast<U4*>(&As[row * 32 + cb + 8]) = d1;
    }
    // ---- stage B tile (128 weight rows x 32 K) into LDS as bf16 ----
    {
      int row = tid >> 1, cb = (tid & 1) * 16;
      const F4s* src = (const F4s*)(Bw + (size_t)(bn + row) * K + k0 + cb);
      F4s f0 = src[0], f1 = src[1], f2 = src[2], f3 = src[3];
      U4 d0 = {pk2(f0.x, f0.y), pk2(f0.z, f0.w), pk2(f1.x, f1.y),
               pk2(f1.z, f1.w)};
      U4 d1 = {pk2(f2.x, f2.y), pk2(f2.z, f2.w), pk2(f3.x, f3.y),
               pk2(f3.z, f3.w)};
      *reinterpret_cast<U4*>(&Bs[row * 32 + cb]) = d0;
      *reinterpret_cast<U4*>(&Bs[row * 32 + cb + 8]) = d1;
      if (k0 + 32 < K)  // pull next weight tile toward L2 (global_prefetch_b8)
        __builtin_prefetch((const void*)(Bw + (size_t)(bn + row) * K + k0 + 32),
                           0, 1);
    }
    if (ABF16 && wave == 0) wait_tensorcnt0();
    __syncthreads();

    v16bf af[2], bf_[4];
    for (int i = 0; i < 2; ++i) af[i] = ld_frag(As, 32, wm + i * 16, 0, lane);
    for (int j = 0; j < 4; ++j) bf_[j] = ld_frag(Bs, 32, wn + j * 16, 0, lane);
    for (int i = 0; i < 2; ++i)
      for (int j = 0; j < 4; ++j) acc[i][j] = wmma_bf16(af[i], bf_[j], acc[i][j]);
    __syncthreads();
  }

  // ---- epilogue: bias + fp32 store (C frag: VGPR r -> M=r / r+8, N=lane&15)
  for (int j = 0; j < 4; ++j) {
    const int col = bn + wn + j * 16 + (lane & 15);
    const float bv = bias[col];
    for (int i = 0; i < 2; ++i) {
      for (int r = 0; r < 8; ++r) {
        const int row = bm + wm + i * 16 + r + ((lane >> 4) << 3);
        C[(size_t)row * N + col] = acc[i][j][r] + bv;
      }
    }
  }
}

// ---------------------------------------------------------------------------
// RoPE (NEOX) + split qkv into bf16 Q[h][t][d], K[kv][t][d], V[kv][t][d].
// grid = (T, H + 2*KV), 64 threads (one per rotation pair).
// ---------------------------------------------------------------------------
__global__ __launch_bounds__(64) void rope_split(
    const int* __restrict__ pos, const float* __restrict__ qkv,
    unsigned short* __restrict__ Qb, unsigned short* __restrict__ Kb,
    unsigned short* __restrict__ Vb) {
  const int t  = blockIdx.x;
  const int hd = blockIdx.y;  // 0..35 q, 36..39 k, 40..43 v
  const int i  = threadIdx.x; // 0..63
  const float p   = (float)pos[t];
  const float inv = __powf(100000.0f, -(float)i / 64.0f);
  float sa, ca;
  __sincosf(p * inv, &sa, &ca);

  if (hd < HEADS) {
    const float* src = qkv + (size_t)t * NQKV + hd * DH;
    unsigned short* dst = Qb + ((size_t)hd * TT + t) * DH;
    float x1 = src[i], x2 = src[i + 64];
    dst[i]      = f2bf(x1 * ca - x2 * sa);
    dst[i + 64] = f2bf(x2 * ca + x1 * sa);
  } else if (hd < HEADS + KVH) {
    const int kh = hd - HEADS;
    const float* src = qkv + (size_t)t * NQKV + HEADS * DH + kh * DH;
    unsigned short* dst = Kb + ((size_t)kh * TT + t) * DH;
    float x1 = src[i], x2 = src[i + 64];
    dst[i]      = f2bf(x1 * ca - x2 * sa);
    dst[i + 64] = f2bf(x2 * ca + x1 * sa);
  } else {
    const int vh = hd - HEADS - KVH;
    const float* src = qkv + (size_t)t * NQKV + (HEADS + KVH) * DH + vh * DH;
    unsigned short* dst = Vb + ((size_t)vh * TT + t) * DH;
    dst[i]      = f2bf(src[i]);
    dst[i + 64] = f2bf(src[i + 64]);
  }
}

// ---------------------------------------------------------------------------
// Flash-attention-2, sliding window.  grid = (T/64, H), 128 threads = 4 waves,
// each wave owns 16 query rows.  Scores and PV via WMMA bf16; K tile staged
// by the Tensor Data Mover; softmax row reductions via DPP16 row rotations.
// ---------------------------------------------------------------------------
__global__ __launch_bounds__(128) void attn_fwd(
    const unsigned short* __restrict__ Qb, const unsigned short* __restrict__ Kb,
    const unsigned short* __restrict__ Vb, unsigned short* __restrict__ Ob,
    const int* __restrict__ wptr) {
  const int h    = blockIdx.y;
  const int qb   = blockIdx.x;
  const int kvh  = h / (HEADS / KVH);
  const int tid  = threadIdx.x, lane = tid & 31, wave = tid >> 5;
  const int window = *wptr;
  const int qrow0  = qb * 64 + wave * 16;

  __shared__ unsigned short Ks[32 * 128];    // [key][d]   (B frag for QK^T)
  __shared__ unsigned short Vs[128 * 32];    // [d][key]   (B frag for PV)
  __shared__ unsigned short Ps[4][16 * 32];  // per-wave P tile for re-frag

  // Q fragments: 4 chunks of K=32 along d, held in registers for the block.
  v16bf qf[4];
  {
    const int row = qrow0 + (lane & 15);
    const int kk  = (lane >> 4) << 3;
    const unsigned short* qp = Qb + ((size_t)h * TT + row) * DH + kk;
    for (int c = 0; c < 4; ++c) {
      F8 f;
      f.lo  = *reinterpret_cast<const U4*>(qp + c * 32);
      f.hi  = *reinterpret_cast<const U4*>(qp + c * 32 + 16);
      qf[c] = __builtin_bit_cast(v16bf, f);
    }
  }

  v8f o[8];
  for (int d = 0; d < 8; ++d) o[d] = vzero();
  float m[8], l[8];
  for (int r = 0; r < 8; ++r) { m[r] = -1e30f; l[r] = 0.f; }

  const int kmin = (qb * 64 - window + 1) > 0 ? (qb * 64 - window + 1) : 0;
  const int sbeg = (kmin >> 5) << 5;
  const int send = qb * 64 + 64;
  const float scale = 0.088388347648318447f;  // 1/sqrt(128)

  for (int s0 = sbeg; s0 < send; s0 += 32) {
    // -- K tile: 32 keys x 128 d, TDM async copy straight into LDS --
    if (wave == 0)
      tdm_load_2d_bf16((unsigned)(uintptr_t)(void*)Ks,
                       Kb + ((size_t)kvh * TT + s0) * DH,
                       /*tile_d0=*/DH, /*tile_d1=*/32, /*row_stride=*/DH);
    // -- V tile load, transposed into Vs[d][key] (TDM cannot transpose) --
    for (int i = 0; i < 4; ++i) {
      int u = tid + i * 128;
      int row = u >> 4, c = (u & 15) * 8;
      U4 v = *reinterpret_cast<const U4*>(Vb + ((size_t)kvh * TT + s0 + row) * DH + c);
      Vs[(c + 0) * 32 + row] = (unsigned short)(v.x);
      Vs[(c + 1) * 32 + row] = (unsigned short)(v.x >> 16);
      Vs[(c + 2) * 32 + row] = (unsigned short)(v.y);
      Vs[(c + 3) * 32 + row] = (unsigned short)(v.y >> 16);
      Vs[(c + 4) * 32 + row] = (unsigned short)(v.z);
      Vs[(c + 5) * 32 + row] = (unsigned short)(v.z >> 16);
      Vs[(c + 6) * 32 + row] = (unsigned short)(v.w);
      Vs[(c + 7) * 32 + row] = (unsigned short)(v.w >> 16);
    }
    if (wave == 0) wait_tensorcnt0();
    __syncthreads();

    // -- scores: two 16x16 tiles (keys s0..15, s0+16..31) --
    v8f sf[2];
    for (int n = 0; n < 2; ++n) {
      v8f s = vzero();
      for (int c = 0; c < 4; ++c) {
        v16bf kf = ld_frag(Ks, 128, n * 16, c * 32, lane);
        s = wmma_bf16(qf[c], kf, s);
      }
      sf[n] = s;
    }

    // -- online softmax; DPP16 rotate-reduce inside each 16-lane row --
    float p0[8], p1[8];
    const int colb = s0 + (lane & 15);
    for (int r = 0; r < 8; ++r) {
      const int row = qrow0 + r + ((lane >> 4) << 3);
      const int c0 = colb, c1 = colb + 16;
      const bool ok0 = (c0 <= row) && (c0 > row - window);
      const bool ok1 = (c1 <= row) && (c1 > row - window);
      const float s0v = ok0 ? sf[0][r] * scale : -1e30f;
      const float s1v = ok1 ? sf[1][r] * scale : -1e30f;
      const float best  = rowmax16(fmaxf(s0v, s1v));
      const float mn    = fmaxf(m[r], best);
      const float alpha = __expf(m[r] - mn);
      m[r] = mn;
      const float e0 = ok0 ? __expf(s0v - mn) : 0.f;
      const float e1 = ok1 ? __expf(s1v - mn) : 0.f;
      p0[r] = e0;
      p1[r] = e1;
      const float rs = rowsum16(e0 + e1);
      l[r] = l[r] * alpha + rs;
      for (int d = 0; d < 8; ++d) o[d][r] *= alpha;
    }

    // -- P (C-frag layout) -> LDS -> re-load as WMMA A fragment --
    unsigned short* pp = &Ps[wave][0];
    {
      const int rlo = (lane >> 4) << 3;
      const int cl  = lane & 15;
      for (int r = 0; r < 8; ++r) {
        pp[(rlo + r) * 32 + cl]      = f2bf(p0[r]);
        pp[(rlo + r) * 32 + cl + 16] = f2bf(p1[r]);
      }
    }
    asm volatile("s_wait_dscnt 0" ::: "memory");  // wave-local RAW through LDS
    v16bf pf = ld_frag(pp, 32, 0, 0, lane);
    for (int d = 0; d < 8; ++d) {
      v16bf vf = ld_frag(Vs, 32, d * 16, 0, lane);
      o[d] = wmma_bf16(pf, vf, o[d]);
    }
    __syncthreads();
  }

  // -- epilogue: normalize, write bf16 attn output [t][h*128 + d] --
  for (int r = 0; r < 8; ++r) l[r] = 1.0f / l[r];
  for (int d = 0; d < 8; ++d) {
    const int col = h * DH + d * 16 + (lane & 15);
    for (int r = 0; r < 8; ++r) {
      const int row = qrow0 + r + ((lane >> 4) << 3);
      Ob[(size_t)row * (HEADS * DH) + col] = f2bf(o[d][r] * l[r]);
    }
  }
}

// ---------------------------------------------------------------------------
extern "C" void kernel_launch(void* const* d_in, const int* in_sizes, int n_in,
                              void* d_out, int out_size, void* d_ws,
                              size_t ws_size, hipStream_t stream) {
  (void)in_sizes; (void)n_in; (void)out_size; (void)ws_size;
  const int*   positions = (const int*)d_in[0];
  const float* hidden    = (const float*)d_in[1];
  const float* wqkv      = (const float*)d_in[2];
  const float* bqkv      = (const float*)d_in[3];
  const float* wo        = (const float*)d_in[4];
  const float* bo        = (const float*)d_in[5];
  const int*   window    = (const int*)d_in[6];
  float* out = (float*)d_out;

  char* ws = (char*)d_ws;
  // layout (bytes):
  float*          qkv = (float*)(ws);                        // T*5632 f32   = 46,137,344
  unsigned short* Qb  = (unsigned short*)(ws + 46137344);    // 36*T*128 bf16 = 18,874,368
  unsigned short* Kb  = (unsigned short*)(ws + 65011712);    // 4*T*128 bf16  =  2,097,152
  unsigned short* Vb  = (unsigned short*)(ws + 67108864);    // 4*T*128 bf16  =  2,097,152
  unsigned short* Ab  = (unsigned short*)(ws + 69206016);    // T*4608 bf16   = 18,874,368

  // 1) qkv = hidden @ wqkv^T + bqkv
  dim3 g1(NQKV / 128, TT / 128);
  gemm_rowB_bias<false><<<g1, dim3(256), 0, stream>>>(hidden, wqkv, bqkv, qkv,
                                                      TT, NQKV, HID);
  // 2) RoPE + split to bf16
  dim3 g2(TT, HEADS + 2 * KVH);
  rope_split<<<g2, dim3(64), 0, stream>>>(positions, qkv, Qb, Kb, Vb);
  // 3) sliding-window flash attention (TDM-fed K tiles)
  dim3 g3(TT / 64, HEADS);
  attn_fwd<<<g3, dim3(128), 0, stream>>>(Qb, Kb, Vb, Ab, window);
  // 4) out = attn @ wo^T + bo  (TDM-fed bf16 A tiles)
  dim3 g4(HID / 128, TT / 128);
  gemm_rowB_bias<true><<<g4, dim3(256), 0, stream>>>(Ab, wo, bo, out, TT, HID,
                                                     HID);
}
